// CausalAttention_223338299734
// MI455X (gfx1250) — matrix operand order repaired and checked by reading, not compiled, not run on hardware
//
#include <hip/hip_runtime.h>
#include <hip/hip_bf16.h>

// ---------------------------------------------------------------------------
// Causal MHA (B=2, T=4096, D=768, H=12, d_head=64), fp32 io.
// cvt(+transpose W)->f16, QKV WMMA-GEMM (async-LDS double-buffered, +bias,
// scatter to Q/K/Vt), flash attention (WMMA QK^T + online softmax + WMMA PV),
// out-proj WMMA-GEMM (+bias, fp32 out).
// ---------------------------------------------------------------------------

typedef _Float16 v8h  __attribute__((ext_vector_type(8)));
typedef _Float16 v16h __attribute__((ext_vector_type(16)));
typedef float    v8f  __attribute__((ext_vector_type(8)));
typedef int      v4i  __attribute__((vector_size(16)));

typedef __attribute__((address_space(1))) v4i* as1_v4i;
typedef __attribute__((address_space(3))) v4i* as3_v4i;

#define Bsz 2
#define Tsz 4096
#define Dm  768
#define Hh  12
#define Dh  64
#define Nq  (3*Dm)        // 2304
#define Mrows (Bsz*Tsz)   // 8192
#define NT  (Dm/32)       // 24 k-steps

#ifndef __has_builtin
#define __has_builtin(x) 0
#endif

#if __has_builtin(__builtin_amdgcn_global_load_async_to_lds_b128)
#define USE_ASYNC_LDS 1
#else
#define USE_ASYNC_LDS 0
#endif

// 16-byte global -> LDS copy (async on CDNA5 when available).
__device__ __forceinline__ void copy16(const _Float16* g, _Float16* l) {
#if USE_ASYNC_LDS
  __builtin_amdgcn_global_load_async_to_lds_b128(
      (as1_v4i)const_cast<_Float16*>(g), (as3_v4i)l, 0, 0);
#else
  *(v8h*)l = *(const v8h*)g;
#endif
}

__device__ __forceinline__ void wait_async_copies() {
#if USE_ASYNC_LDS
  asm volatile("s_wait_asynccnt 0" ::: "memory");
#endif
}

__device__ __forceinline__ v16h make_frag(const _Float16* p0, const _Float16* p1) {
  v8h a = *(const v8h*)p0;
  v8h b = *(const v8h*)p1;
  v16h r;
#pragma unroll
  for (int i = 0; i < 8; ++i) { r[i] = a[i]; r[8 + i] = b[i]; }
  return r;
}

__device__ __forceinline__ v8f wmma_f16(v16h a, v16h b, v8f c) {
  return __builtin_amdgcn_wmma_f32_16x16x32_f16(false, a, false, b, (short)0, c,
                                                false, false);
}

// ------------------------------- f32 -> f16 --------------------------------
__global__ void cvt_f16_kernel(const float* __restrict__ in,
                               _Float16* __restrict__ out, int n) {
  int i = blockIdx.x * blockDim.x + threadIdx.x;
  if (i < n) out[i] = (_Float16)in[i];
}

// f32 [K][N] -> f16 transposed [N][K] (coalesced writes; one-time cost).
__global__ void cvt_f16_t_kernel(const float* __restrict__ in,
                                 _Float16* __restrict__ out, int K, int N) {
  int i = blockIdx.x * blockDim.x + threadIdx.x;
  if (i < K * N) {
    int n = i / K, k = i - n * K;
    out[i] = (_Float16)in[(size_t)k * N + n];
  }
}

// ----------------------- QKV GEMM: [8192x768]x[768x2304] -------------------
// A row-major [M][K] f16, Wt transposed [N][K] f16. Block 256 thr = 8 waves
// (2x4). Tile 128x128, K-step 32, double-buffered async LDS staging.
__global__ __launch_bounds__(256) void qkv_gemm_kernel(
    const _Float16* __restrict__ A, const _Float16* __restrict__ Wt,
    const float* __restrict__ bias, _Float16* __restrict__ Qo,
    _Float16* __restrict__ Ko, _Float16* __restrict__ Vto) {
  __shared__ _Float16 As[2][128][32];
  __shared__ _Float16 Bt[2][128][32];   // [n][k]
  const int tid = threadIdx.x;
  const int lane = tid & 31, wid = tid >> 5;
  const int lmod = lane & 15, lhalf = lane >> 4;
  const int wm = wid & 1, wn = wid >> 1;         // 2 x 4 waves
  const int m0 = blockIdx.x * 128, n0 = blockIdx.y * 128;
  const int cm = tid >> 2, ck = (tid & 3) * 8;   // per-thread copy coords

  v8f acc[4][2] = {};

  // Branchless: each thread moves 4x16B per tile (A rows cm, cm+64; B same).
  auto issue_tile = [&](int kt, int buf) {
    const int k0 = kt * 32;
    copy16(A  + (size_t)(m0 + cm)      * Dm + k0 + ck, &As[buf][cm][ck]);
    copy16(A  + (size_t)(m0 + cm + 64) * Dm + k0 + ck, &As[buf][cm + 64][ck]);
    copy16(Wt + (size_t)(n0 + cm)      * Dm + k0 + ck, &Bt[buf][cm][ck]);
    copy16(Wt + (size_t)(n0 + cm + 64) * Dm + k0 + ck, &Bt[buf][cm + 64][ck]);
  };

  auto compute_tile = [&](int cur) {
    v16h af[4], bf[2];
#pragma unroll
    for (int sm = 0; sm < 4; ++sm) {
      const _Float16* p = &As[cur][wm * 64 + sm * 16 + lmod][16 * lhalf];
      af[sm] = make_frag(p, p + 8);
    }
#pragma unroll
    for (int sn = 0; sn < 2; ++sn) {
      const _Float16* p = &Bt[cur][wn * 32 + sn * 16 + lmod][16 * lhalf];
      bf[sn] = make_frag(p, p + 8);
    }
#pragma unroll
    for (int sm = 0; sm < 4; ++sm)
#pragma unroll
      for (int sn = 0; sn < 2; ++sn)
        acc[sm][sn] = wmma_f16(af[sm], bf[sn], acc[sm][sn]);
  };

  issue_tile(0, 0);
#pragma unroll 1
  for (int kt = 0; kt < NT - 1; ++kt) {
    const int cur = kt & 1;
    wait_async_copies();
    __syncthreads();                       // tile kt visible to all waves
    issue_tile(kt + 1, cur ^ 1);
    compute_tile(cur);
  }
  wait_async_copies();
  __syncthreads();
  compute_tile((NT - 1) & 1);

#pragma unroll
  for (int sm = 0; sm < 4; ++sm)
#pragma unroll
    for (int sn = 0; sn < 2; ++sn)
#pragma unroll
      for (int j = 0; j < 8; ++j) {
        int row = m0 + wm * 64 + sm * 16 + j + 8 * lhalf;
        int col = n0 + wn * 32 + sn * 16 + lmod;
        _Float16 hv = (_Float16)(acc[sm][sn][j] + bias[col]);
        int bb = row >> 12, t = row & 4095;
        if (col < Dm) {
          int h = col >> 6, d = col & 63;
          Qo[(((size_t)bb * Hh + h) * Tsz + t) * Dh + d] = hv;
        } else if (col < 2 * Dm) {
          int c = col - Dm, h = c >> 6, d = c & 63;
          Ko[(((size_t)bb * Hh + h) * Tsz + t) * Dh + d] = hv;
        } else {
          int c = col - 2 * Dm, h = c >> 6, d = c & 63;
          Vto[(((size_t)bb * Hh + h) * Dh + d) * Tsz + t] = hv;
        }
      }
}

// ------------------------------ Flash attention ----------------------------
// Block = 128 thr = 4 waves; each wave owns 16 query rows of one (b,h).
// Key tiles of 32: S(16x32)=Q(16x64)K^T (4 wmma), online softmax f32,
// P via per-wave LDS relayout, O += P(16x32)V(32x64) (4 wmma).
__global__ __launch_bounds__(128) void flash_kernel(
    const _Float16* __restrict__ Q, const _Float16* __restrict__ K,
    const _Float16* __restrict__ Vt, _Float16* __restrict__ AO) {
  const int bh = blockIdx.x;                 // 0..23
  const int b = bh / Hh, h = bh % Hh;
  const int wid = threadIdx.x >> 5;
  const int lane = threadIdx.x & 31;
  const int lmod = lane & 15, lhalf = lane >> 4;
  const int qbase = blockIdx.y * 64 + wid * 16;

  const _Float16* Qh = Q + (size_t)bh * Tsz * Dh;
  const _Float16* Kh = K + (size_t)bh * Tsz * Dh;
  const _Float16* Vh = Vt + (size_t)bh * Dh * Tsz;

  __shared__ _Float16 Plds[4][16][32];
  _Float16(*Pw)[32] = Plds[wid];

  // Q fragments (A layout): row = qbase+lmod, contiguous 8-elem K runs.
  v16h qf[2];
  {
    const _Float16* qrow = Qh + (size_t)(qbase + lmod) * Dh;
    qf[0] = make_frag(qrow + 16 * lhalf, qrow + 8 + 16 * lhalf);       // d 0..31
    qf[1] = make_frag(qrow + 32 + 16 * lhalf, qrow + 40 + 16 * lhalf);  // d 32..63
  }

  v8f o[4] = {};
  float mrow[8], lrow[8];
#pragma unroll
  for (int j = 0; j < 8; ++j) { mrow[j] = -1e30f; lrow[j] = 0.0f; }

  const float scale = 0.125f;                // 1/sqrt(64)
  const int ntiles = (qbase + 16 + 31) / 32;

  for (int t = 0; t < ntiles; ++t) {
    const int k0 = t * 32;
    v8f s0 = {}, s1 = {};
#pragma unroll
    for (int kc = 0; kc < 2; ++kc) {
      // B frag of K^T: lane = key col, 16 contiguous dims per lane.
      const _Float16* kr0 = Kh + (size_t)(k0 + lmod) * Dh + kc * 32 + 16 * lhalf;
      const _Float16* kr1 = Kh + (size_t)(k0 + 16 + lmod) * Dh + kc * 32 + 16 * lhalf;
      s0 = wmma_f16(qf[kc], make_frag(kr0, kr0 + 8), s0);
      s1 = wmma_f16(qf[kc], make_frag(kr1, kr1 + 8), s1);
    }
#pragma unroll
    for (int j = 0; j < 8; ++j) {
      const int qg = qbase + j + 8 * lhalf;
      float v0 = s0[j] * scale;
      float v1 = s1[j] * scale;
      if (k0 + lmod > qg)      v0 = -1e30f;
      if (k0 + 16 + lmod > qg) v1 = -1e30f;
      float mx = fmaxf(v0, v1);
#pragma unroll
      for (int s = 1; s < 16; s <<= 1) mx = fmaxf(mx, __shfl_xor(mx, s, 32));
      const float mnew = fmaxf(mrow[j], mx);
      const float r = __expf(mrow[j] - mnew);
      const float p0 = __expf(v0 - mnew);
      const float p1 = __expf(v1 - mnew);
      float ps = p0 + p1;
#pragma unroll
      for (int s = 1; s < 16; s <<= 1) ps += __shfl_xor(ps, s, 32);
      lrow[j] = lrow[j] * r + ps;
      mrow[j] = mnew;
#pragma unroll
      for (int d = 0; d < 4; ++d) o[d][j] *= r;
      Pw[j + 8 * lhalf][lmod] = (_Float16)p0;
      Pw[j + 8 * lhalf][16 + lmod] = (_Float16)p1;
    }
    asm volatile("s_wait_dscnt 0" ::: "memory");   // same-wave LDS transpose
    v16h pf = make_frag(&Pw[lmod][16 * lhalf], &Pw[lmod][8 + 16 * lhalf]);
#pragma unroll
    for (int dsub = 0; dsub < 4; ++dsub) {
      const _Float16* vp = Vh + (size_t)(dsub * 16 + lmod) * Tsz + k0 + 16 * lhalf;
      o[dsub] = wmma_f16(pf, make_frag(vp, vp + 8), o[dsub]);
    }
  }

#pragma unroll
  for (int dsub = 0; dsub < 4; ++dsub)
#pragma unroll
    for (int j = 0; j < 8; ++j) {
      const int qg = qbase + j + 8 * lhalf;
      float val = o[dsub][j] / lrow[j];
      AO[(size_t)(b * Tsz + qg) * Dm + h * Dh + dsub * 16 + lmod] = (_Float16)val;
    }
}

// --------------------- Out-proj GEMM: [8192x768]x[768x768] -----------------
__global__ __launch_bounds__(256) void out_gemm_kernel(
    const _Float16* __restrict__ A, const _Float16* __restrict__ Wt,
    const float* __restrict__ bias, float* __restrict__ Co) {
  __shared__ _Float16 As[2][128][32];
  __shared__ _Float16 Bt[2][128][32];
  const int tid = threadIdx.x;
  const int lane = tid & 31, wid = tid >> 5;
  const int lmod = lane & 15, lhalf = lane >> 4;
  const int wm = wid & 1, wn = wid >> 1;
  const int m0 = blockIdx.x * 128, n0 = blockIdx.y * 128;
  const int cm = tid >> 2, ck = (tid & 3) * 8;

  v8f acc[4][2] = {};

  auto issue_tile = [&](int kt, int buf) {
    const int k0 = kt * 32;
    copy16(A  + (size_t)(m0 + cm)      * Dm + k0 + ck, &As[buf][cm][ck]);
    copy16(A  + (size_t)(m0 + cm + 64) * Dm + k0 + ck, &As[buf][cm + 64][ck]);
    copy16(Wt + (size_t)(n0 + cm)      * Dm + k0 + ck, &Bt[buf][cm][ck]);
    copy16(Wt + (size_t)(n0 + cm + 64) * Dm + k0 + ck, &Bt[buf][cm + 64][ck]);
  };

  auto compute_tile = [&](int cur) {
    v16h af[4], bf[2];
#pragma unroll
    for (int sm = 0; sm < 4; ++sm) {
      const _Float16* p = &As[cur][wm * 64 + sm * 16 + lmod][16 * lhalf];
      af[sm] = make_frag(p, p + 8);
    }
#pragma unroll
    for (int sn = 0; sn < 2; ++sn) {
      const _Float16* p = &Bt[cur][wn * 32 + sn * 16 + lmod][16 * lhalf];
      bf[sn] = make_frag(p, p + 8);
    }
#pragma unroll
    for (int sm = 0; sm < 4; ++sm)
#pragma unroll
      for (int sn = 0; sn < 2; ++sn)
        acc[sm][sn] = wmma_f16(af[sm], bf[sn], acc[sm][sn]);
  };

  issue_tile(0, 0);
#pragma unroll 1
  for (int kt = 0; kt < NT - 1; ++kt) {
    const int cur = kt & 1;
    wait_async_copies();
    __syncthreads();
    issue_tile(kt + 1, cur ^ 1);
    compute_tile(cur);
  }
  wait_async_copies();
  __syncthreads();
  compute_tile((NT - 1) & 1);

#pragma unroll
  for (int sm = 0; sm < 4; ++sm)
#pragma unroll
    for (int sn = 0; sn < 2; ++sn)
#pragma unroll
      for (int j = 0; j < 8; ++j) {
        int row = m0 + wm * 64 + sm * 16 + j + 8 * lhalf;
        int col = n0 + wn * 32 + sn * 16 + lmod;
        Co[(size_t)row * Dm + col] = acc[sm][sn][j] + bias[col];
      }
}

// ------------------------------- launcher ----------------------------------
extern "C" void kernel_launch(void* const* d_in, const int* in_sizes, int n_in,
                              void* d_out, int out_size, void* d_ws, size_t ws_size,
                              hipStream_t stream) {
  const float* x     = (const float*)d_in[0];   // [2,4096,768]
  const float* Wqkv  = (const float*)d_in[1];   // [768,2304]
  const float* bqkv  = (const float*)d_in[2];   // [2304]
  const float* Wout  = (const float*)d_in[3];   // [768,768]
  const float* bout  = (const float*)d_in[4];   // [768]
  float* out = (float*)d_out;

  char* ws = (char*)d_ws;
  size_t off = 0;
  auto carve = [&](size_t bytes) {
    void* p = ws + off;
    off += (bytes + 255) & ~(size_t)255;
    return p;
  };
  _Float16* xh     = (_Float16*)carve((size_t)Mrows * Dm * 2);
  _Float16* Wqkvt  = (_Float16*)carve((size_t)Dm * Nq * 2);   // [2304][768]
  _Float16* Woutt  = (_Float16*)carve((size_t)Dm * Dm * 2);   // [768][768] T
  _Float16* Qf     = (_Float16*)carve((size_t)Bsz * Hh * Tsz * Dh * 2);
  _Float16* Kf     = (_Float16*)carve((size_t)Bsz * Hh * Tsz * Dh * 2);
  _Float16* Vtf    = (_Float16*)carve((size_t)Bsz * Hh * Dh * Tsz * 2);
  _Float16* AO     = (_Float16*)carve((size_t)Mrows * Dm * 2);
  (void)ws_size;

  {
    int n = Mrows * Dm;
    cvt_f16_kernel<<<(n + 255) / 256, 256, 0, stream>>>(x, xh, n);
  }
  {
    int n = Dm * Nq;
    cvt_f16_t_kernel<<<(n + 255) / 256, 256, 0, stream>>>(Wqkv, Wqkvt, Dm, Nq);
  }
  {
    int n = Dm * Dm;
    cvt_f16_t_kernel<<<(n + 255) / 256, 256, 0, stream>>>(Wout, Woutt, Dm, Dm);
  }

  qkv_gemm_kernel<<<dim3(Mrows / 128, Nq / 128), 256, 0, stream>>>(
      xh, Wqkvt, bqkv, Qf, Kf, Vtf);

  flash_kernel<<<dim3(Bsz * Hh, Tsz / 64), 128, 0, stream>>>(Qf, Kf, Vtf, AO);

  out_gemm_kernel<<<dim3(Mrows / 128, Dm / 128), 256, 0, stream>>>(
      AO, Woutt, bout, out);
}
